// QuGCN_54202487276109
// MI455X (gfx1250) — matrix operand order repaired and checked by reading, not compiled, not run on hardware
//
#include <hip/hip_runtime.h>
#include <hip/hip_bf16.h>

typedef __attribute__((ext_vector_type(16))) _Float16 v16h;
typedef __attribute__((ext_vector_type(8)))  float    v8f;
typedef __attribute__((ext_vector_type(4)))  int      v4i;

#define NW 132

#if __has_builtin(__builtin_amdgcn_global_load_async_to_lds_b128) && \
    __has_builtin(__builtin_amdgcn_s_wait_asynccnt)
#define USE_ASYNC_LDS 1
typedef __attribute__((address_space(1))) v4i gv4i;   // global int4
typedef __attribute__((address_space(3))) v4i lv4i;   // LDS int4
#endif

// ---------------- gate matrices (2x2 complex, row major) ----------------
struct C2 { float ar,ai,br,bi,cr,ci,dr,di; };

__device__ __forceinline__ C2 gRY(float t){ float c=cosf(0.5f*t), s=sinf(0.5f*t);
  return C2{c,0.f,-s,0.f,  s,0.f, c,0.f}; }
__device__ __forceinline__ C2 gRX(float t){ float c=cosf(0.5f*t), s=sinf(0.5f*t);
  return C2{c,0.f, 0.f,-s, 0.f,-s, c,0.f}; }
__device__ __forceinline__ C2 gRZ(float t){ float c=cosf(0.5f*t), s=sinf(0.5f*t);
  return C2{c,-s, 0.f,0.f, 0.f,0.f, c, s}; }

// control bit for element (register r, lane); cw<0 => uncontrolled
__device__ __forceinline__ bool ctrl_on(int cw, int r, int lane){
  if (cw < 0) return true;
  int s = 1 << (7 - cw);                       // wire stride (wire 0 = MSB)
  return (s >= 32) ? ((r & (s >> 5)) != 0) : ((lane & s) != 0);
}

// Apply 2x2 complex gate g on target wire tw (control cw) to the distributed
// state: amplitude n = r*32 + lane lives in sr[r],si[r] of lane.
__device__ __forceinline__ void applyGate(const C2 g, int tw, int cw,
                                          float sr[8], float si[8], int lane)
{
  int s = 1 << (7 - tw);
  if (s >= 32) {                               // register-bit pairing
    int mr = s >> 5;                           // 1,2,4
#define PAIR_OP(R0, R1) {                                                  \
      bool on = ctrl_on(cw, R0, lane);                                     \
      float v0r=sr[R0], v0i=si[R0], v1r=sr[R1], v1i=si[R1];                \
      float n0r = g.ar*v0r - g.ai*v0i + g.br*v1r - g.bi*v1i;               \
      float n0i = g.ar*v0i + g.ai*v0r + g.br*v1i + g.bi*v1r;               \
      float n1r = g.cr*v0r - g.ci*v0i + g.dr*v1r - g.di*v1i;               \
      float n1i = g.cr*v0i + g.ci*v0r + g.dr*v1i + g.di*v1r;               \
      if (on){ sr[R0]=n0r; si[R0]=n0i; sr[R1]=n1r; si[R1]=n1i; } }
    if      (mr == 1) { PAIR_OP(0,1) PAIR_OP(2,3) PAIR_OP(4,5) PAIR_OP(6,7) }
    else if (mr == 2) { PAIR_OP(0,2) PAIR_OP(1,3) PAIR_OP(4,6) PAIR_OP(5,7) }
    else              { PAIR_OP(0,4) PAIR_OP(1,5) PAIR_OP(2,6) PAIR_OP(3,7) }
#undef PAIR_OP
  } else {                                     // lane-bit pairing (shfl_xor)
    bool hi = (lane & s) != 0;
#pragma unroll
    for (int r = 0; r < 8; ++r) {
      float pr = __shfl_xor(sr[r], s, 32);
      float pi = __shfl_xor(si[r], s, 32);
      bool on = ctrl_on(cw, r, lane);
      float nr, ni;
      if (hi) { nr = g.cr*pr - g.ci*pi + g.dr*sr[r] - g.di*si[r];
                ni = g.cr*pi + g.ci*pr + g.dr*si[r] + g.di*sr[r]; }
      else    { nr = g.ar*sr[r] - g.ai*si[r] + g.br*pr - g.bi*pi;
                ni = g.ar*si[r] + g.ai*sr[r] + g.br*pi + g.bi*pr; }
      if (on) { sr[r] = nr; si[r] = ni; }
    }
  }
}

// ---------------- Kernel A: build U_total by evolving 256 basis states ----
// Wave (block of 32) k evolves e_k through all 4 layers; writes transposed
// f16 matrices Bt[n][k] = U[n][k] (real & imag) so kernel B's WMMA B-fragment
// is a contiguous 16-half load per lane.
__global__ void build_unitary_k(const float* __restrict__ qw,
                                _Float16* __restrict__ btr,
                                _Float16* __restrict__ bti)
{
  const int lane = threadIdx.x;
  const int k = blockIdx.x;
  float sr[8], si[8];
#pragma unroll
  for (int r = 0; r < 8; ++r) { sr[r] = ((r*32 + lane) == k) ? 1.f : 0.f; si[r] = 0.f; }

  for (int l = 0; l < 4; ++l) {
    const float* wv = qw + l*NW;
    int wi = 0;
    for (int q = 0; q < 8; ++q) applyGate(gRY(wv[wi+q]), q, -1, sr, si, lane);
    wi += 8;
    { int idx = 0;                             // CRY, angle == weight index (reference quirk)
      for (int gap = 1; gap < 8; ++gap)
        for (int j = 0; j + gap < 8; ++j)
          applyGate(gRY((float)(wi + idx++)), j+gap, j, sr, si, lane); }
    wi += 28;
    for (int q = 0; q < 8; ++q) applyGate(gRY(wv[wi+q]), q, -1, sr, si, lane);
    wi += 8;
    for (int q = 0; q < 8; ++q) applyGate(gRX(wv[wi+q]), q, -1, sr, si, lane);
    wi += 8;
    { int idx = 0;                             // CRX
      for (int gap = 1; gap < 8; ++gap)
        for (int j = 0; j + gap < 8; ++j)
          applyGate(gRX((float)(wi + idx++)), j+gap, j, sr, si, lane); }
    wi += 28;
    for (int q = 0; q < 8; ++q) applyGate(gRX(wv[wi+q]), q, -1, sr, si, lane);
    wi += 8;
    for (int q = 0; q < 8; ++q) applyGate(gRZ(wv[wi+q]), q, -1, sr, si, lane);
    wi += 8;
    { int idx = 0;                             // CRZ
      for (int gap = 1; gap < 8; ++gap)
        for (int j = 0; j + gap < 8; ++j)
          applyGate(gRZ((float)(wi + idx++)), j+gap, j, sr, si, lane); }
    wi += 28;
    for (int q = 0; q < 8; ++q) applyGate(gRZ(wv[wi+q]), q, -1, sr, si, lane);
  }

#pragma unroll
  for (int r = 0; r < 8; ++r) {
    int n = r*32 + lane;
    btr[n*256 + k] = (_Float16)sr[r];
    bti[n*256 + k] = (_Float16)si[r];
  }
}

// ---------------- Kernel B: WMMA GEMM + measurement epilogue --------------
// Block = one 16-row batch tile (M). 8 waves; wave w computes N-columns
// 32w..32w+31 over the full K=256. S = A * (Wr + i Wi), A real (f16 from LDS),
// accumulate f32. Then p = Sr^2 + Si^2, z/FC head folded into one reduction.
__global__ __launch_bounds__(256)
void qgemm_k(const float* __restrict__ x,
             const float* __restrict__ fcw,
             const float* __restrict__ fcb,
             const _Float16* __restrict__ btr,
             const _Float16* __restrict__ bti,
             float* __restrict__ out)
{
  __shared__ float A_lds[256 * 16];            // [k][b] f32, 16KB
  __shared__ float p_lds[16 * 256];            // [b][n] f32, 16KB

  const int t  = threadIdx.x;
  const int m0 = blockIdx.x * 16;

  // ---- stage A block: A[b,k] = x[k, m0+b]; row k of x is contiguous in b
#ifdef USE_ASYNC_LDS
  { // CDNA5 async global->LDS (no VGPR bounce), tracked by ASYNCcnt
    const float* src = x + (size_t)t*4096 + m0;
#pragma unroll
    for (int j = 0; j < 4; ++j) {
      __builtin_amdgcn_global_load_async_to_lds_b128(
          (gv4i*)(src + 4*j),
          (lv4i*)(A_lds + t*16 + 4*j),
          0, 0);
    }
    __builtin_amdgcn_s_wait_asynccnt(0);
  }
#else
  {
    const float4* src = (const float4*)(x + (size_t)t*4096 + m0);
#pragma unroll
    for (int j = 0; j < 4; ++j) {
      float4 v = src[j];
      A_lds[t*16 + 4*j + 0] = v.x; A_lds[t*16 + 4*j + 1] = v.y;
      A_lds[t*16 + 4*j + 2] = v.z; A_lds[t*16 + 4*j + 3] = v.w;
    }
  }
#endif
  __syncthreads();

  const int wave = t >> 5, lane = t & 31;
  v8f cr0 = {}, ci0 = {}, cr1 = {}, ci1 = {};
  const int m  = lane & 15;                    // A-fragment M row
  const int n0 = wave*32 + (lane & 15);        // B-fragment N column (tile 0)

  const _Float16* pr0 = btr + (size_t)n0*256;
  const _Float16* pi0 = bti + (size_t)n0*256;
  const _Float16* pr1 = btr + (size_t)(n0+16)*256;
  const _Float16* pi1 = bti + (size_t)(n0+16)*256;
  const int kb = (lane & 16) ? 16 : 0;         // K offset for this half-wave

  // prime double buffer with ks = 0 fragments
  v16h br0c = *(const v16h*)(pr0 + kb);
  v16h bi0c = *(const v16h*)(pi0 + kb);
  v16h br1c = *(const v16h*)(pr1 + kb);
  v16h bi1c = *(const v16h*)(pi1 + kb);

#pragma unroll
  for (int ks = 0; ks < 8; ++ks) {
    // issue next k-step's B fragments before consuming the current ones
    v16h br0n, bi0n, br1n, bi1n;
    if (ks < 7) {
      int kn = (ks + 1)*32 + kb;
      br0n = *(const v16h*)(pr0 + kn);
      bi0n = *(const v16h*)(pi0 + kn);
      br1n = *(const v16h*)(pr1 + kn);
      bi1n = *(const v16h*)(pi1 + kn);
    }

    // A fragment (contiguous K, matches B): lanes 0-15: K=+0..15;
    // lanes 16-31: K=+16..31
    v16h a;
    int klo = ks*32 + ((lane & 16) ? 16 : 0);
#pragma unroll
    for (int j = 0; j < 8; ++j) a[j]     = (_Float16)A_lds[(klo + j     )*16 + m];
#pragma unroll
    for (int j = 0; j < 8; ++j) a[j + 8] = (_Float16)A_lds[(klo + 8 + j)*16 + m];

    cr0 = __builtin_amdgcn_wmma_f32_16x16x32_f16(false, a, false, br0c, (short)0, cr0, false, false);
    ci0 = __builtin_amdgcn_wmma_f32_16x16x32_f16(false, a, false, bi0c, (short)0, ci0, false, false);
    cr1 = __builtin_amdgcn_wmma_f32_16x16x32_f16(false, a, false, br1c, (short)0, cr1, false, false);
    ci1 = __builtin_amdgcn_wmma_f32_16x16x32_f16(false, a, false, bi1c, (short)0, ci1, false, false);

    if (ks < 7) { br0c = br0n; bi0c = bi0n; br1c = br1n; bi1c = bi1n; }
  }

  { // probabilities -> LDS (C/D layout: lanes 0-15: M=r; lanes 16-31: M=8+r)
    int nc0 = wave*32 + (lane & 15);
    int mb  = (lane & 16) ? 8 : 0;
#pragma unroll
    for (int r = 0; r < 8; ++r) {
      p_lds[(mb + r)*256 + nc0     ] = cr0[r]*cr0[r] + ci0[r]*ci0[r];
      p_lds[(mb + r)*256 + nc0 + 16] = cr1[r]*cr1[r] + ci1[r]*ci1[r];
    }
  }
  __syncthreads();

  // Epilogue: out[b,o] = sum_i G[o,i] * p[b,i] / ||x_b||^2 + fc_b[o],
  // G[o,i] = sum_q fc_w[o,q] * (bit_{7-q}(i) ? -1 : +1)
  if (t < 112) {
    const int o = t >> 4, b = t & 15;
    float n2 = 0.f;
    for (int k = 0; k < 256; ++k) { float v = A_lds[k*16 + b]; n2 += v*v; }
    float acc = 0.f;
    float w[8];
#pragma unroll
    for (int q = 0; q < 8; ++q) w[q] = fcw[o*8 + q];
    for (int i = 0; i < 256; ++i) {
      float g = 0.f;
#pragma unroll
      for (int q = 0; q < 8; ++q)
        g += ((i >> (7 - q)) & 1) ? -w[q] : w[q];
      acc += g * p_lds[b*256 + i];
    }
    out[(size_t)(m0 + b)*7 + o] = acc / n2 + fcb[o];
  }
}

extern "C" void kernel_launch(void* const* d_in, const int* in_sizes, int n_in,
                              void* d_out, int out_size, void* d_ws, size_t ws_size,
                              hipStream_t stream) {
  const float* x   = (const float*)d_in[0];   // [256, 4096]
  const float* qw  = (const float*)d_in[1];   // [4, 132]
  const float* fcw = (const float*)d_in[2];   // [7, 8]
  const float* fcb = (const float*)d_in[3];   // [7]
  float* out = (float*)d_out;                 // [4096, 7]

  _Float16* btr = (_Float16*)d_ws;            // 256*256 f16 (128KB)
  _Float16* bti = btr + 256*256;              // 256*256 f16 (128KB)

  build_unitary_k<<<256, 32, 0, stream>>>(qw, btr, bti);
  qgemm_k<<<256, 256, 0, stream>>>(x, fcw, fcb, btr, bti, out);
}
